// ConvKAN_33389075759112
// MI455X (gfx1250) — matrix operand order repaired and checked by reading, not compiled, not run on hardware
//
#include <hip/hip_runtime.h>
#include <hip/hip_bf16.h>
#include <stdint.h>

// ---- problem constants ----
constexpr int C_IN   = 64;
constexpr int C_OUT  = 128;
constexpr int HW     = 48;
constexpr int LSPAT  = HW * HW;        // 2304
constexpr int IN_DIM = 576;            // 64*3*3
constexpr int KPE    = 9;              // silu + 8 spline bases per patch element
constexpr int KC     = 32;             // patch elements per K-chunk
constexpr int KSUB   = KC * KPE;       // 288 bf16 K per chunk
constexpr int NCHUNK = IN_DIM / KC;    // 18
constexpr int KTOT   = IN_DIM * KPE;   // 5184
constexpr int MT     = 128;            // rows per workgroup
constexpr int LDW    = 148;            // LDS row stride in dwords (144 used + 4 pad)
constexpr int BROWS  = LSPAT;          // rows per batch image = 2304

typedef __attribute__((ext_vector_type(16))) __bf16        v16bf;
typedef __attribute__((ext_vector_type(2)))  __bf16        v2bf;
typedef __attribute__((ext_vector_type(8)))  float         v8f;
typedef __attribute__((ext_vector_type(4)))  unsigned int  v4u;
typedef __attribute__((ext_vector_type(8)))  unsigned int  v8u;

__device__ __forceinline__ unsigned int pack2(float a, float b) {
  v2bf p;
  p[0] = (__bf16)a;       // RNE f32->bf16; packable to v_cvt_pk if available
  p[1] = (__bf16)b;
  return __builtin_bit_cast(unsigned int, p);
}
__device__ __forceinline__ unsigned short f2bfu(float f) {
  __bf16 h = (__bf16)f;
  return __builtin_bit_cast(unsigned short, h);
}

__device__ __forceinline__ v16bf load_frag(const unsigned int* p0, const unsigned int* p1) {
  v4u lo = *(const v4u*)p0;                 // ds_load_b128
  v4u hi = *(const v4u*)p1;                 // ds_load_b128
  v8u cat = __builtin_shufflevector(lo, hi, 0, 1, 2, 3, 4, 5, 6, 7);
  return __builtin_bit_cast(v16bf, cat);
}

// Async global->LDS 16B copy (GVS mode: SGPR64 base + VGPR 32-bit offset).
__device__ __forceinline__ void async_copy_b128(unsigned int lds_byte_off,
                                                const void* sbase,
                                                unsigned int voff_bytes) {
  asm volatile("global_load_async_to_lds_b128 %0, %1, %2"
               :
               : "v"(lds_byte_off), "v"(voff_bytes), "s"(sbase)
               : "memory");
}
__device__ __forceinline__ void wait_asynccnt0() {
  asm volatile("s_wait_asynccnt 0" ::: "memory");
}

// silu + 8 cubic B-spline bases (uniform grid t_i = -2.2 + 0.4*i).
// Bases are produced scaled by 6; the 1/6 is folded into the spline weights.
// N_s(v)*6 = K3(y), y=(v+2.2)/0.4 - 2 - s:
//   K3 = 4 - 6y^2 + 3|y|^3      for |y|<1
//   K3 = max(2-|y|, 0)^3        otherwise (zero tail included)
__device__ __forceinline__ void kan_feats(float v, float out[KPE]) {
  out[0] = v / (1.0f + __expf(-v));                 // silu
  float y0 = __builtin_fmaf(v, 2.5f, 3.5f);         // (v+2.2)/h - 2
#pragma unroll
  for (int s = 0; s < 8; ++s) {
    float y  = y0 - (float)s;
    float a  = __builtin_fabsf(y);
    float a2 = y * y;
    float a3 = a2 * a;
    float q  = __builtin_fmaxf(2.0f - a, 0.0f);
    float q3 = (q * q) * q;
    float inner = __builtin_fmaf(3.0f, a3, __builtin_fmaf(-6.0f, a2, 4.0f));
    out[1 + s] = (a < 1.0f) ? inner : q3;
  }
}

// Fold base_weight / spline_weight*scaler/6 into one bf16 matrix W[o][kg],
// kg = (j/32)*288 + s*32 + (j%32), s=0 -> base, s=1..8 -> spline basis s-1.
__global__ void kan_prep_w(const float* __restrict__ bw, const float* __restrict__ sw,
                           const float* __restrict__ sc, unsigned short* __restrict__ W) {
  int i = blockIdx.x * 256 + threadIdx.x;
  if (i >= C_OUT * KTOT) return;
  int o  = i / KTOT;
  int k  = i - o * KTOT;
  int jc = k / KSUB;
  int r  = k - jc * KSUB;
  int s  = r >> 5;
  int jl = r & 31;
  int j  = jc * KC + jl;
  float w;
  if (s == 0) w = bw[o * IN_DIM + j];
  else        w = sw[(o * IN_DIM + j) * 8 + (s - 1)] * sc[o * IN_DIM + j] * (1.0f / 6.0f);
  W[i] = f2bfu(w);
}

__global__ __launch_bounds__(256)
void kan_conv_wmma(const float* __restrict__ x, const unsigned int* __restrict__ Wdw,
                   float* __restrict__ out) {
  extern __shared__ unsigned int smem[];
  unsigned int* Adw = smem;                 // 128 rows x 148 dw (bf16 pairs, k = s*32+j)
  unsigned int* Bdw = smem + MT * LDW;      // 128 cols x 148 dw
  const unsigned int BOFF = MT * LDW * 4;   // byte offset of B region in LDS

  const int tid  = threadIdx.x;
  const int lane = tid & 31;
  const int wv   = tid >> 5;
  const int m0   = blockIdx.x * MT;
  const int bidx = m0 / BROWS;
  const int rm0  = m0 - bidx * BROWS;
  const float* xb = x + bidx * (C_IN * LSPAT);

  const int wr   = wv >> 1;     // row-block pair 2wr, 2wr+1
  const int wc   = wv & 1;      // col blocks 4wc .. 4wc+3
  const int half = lane >> 4;
  const int lr   = lane & 15;

  v8f acc[2][4];
#pragma unroll
  for (int i = 0; i < 2; ++i)
#pragma unroll
    for (int jn = 0; jn < 4; ++jn) acc[i][jn] = {};

#pragma unroll 1
  for (int jc = 0; jc < NCHUNK; ++jc) {
    // ---- stage B chunk via async global->LDS DMA (overlaps A-fill VALU) ----
    // 128 cols x 288 bf16 = 4608 16B segments; 18 per thread.
#pragma unroll
    for (int it = 0; it < 18; ++it) {
      int seg = it * 256 + tid;
      int o   = seg / 36;
      int sg  = seg - o * 36;
      unsigned int voff = (unsigned int)(o * (KTOT / 2) + jc * (KSUB / 2) + sg * 4) * 4u;
      unsigned int loff = BOFF + (unsigned int)(o * LDW + sg * 4) * 4u;
      async_copy_b128(loff, Wdw, voff);
    }
    // prefetch next weight chunk into caches (global_prefetch_b8)
    if (jc + 1 < NCHUNK) {
      int o = tid >> 1;
      const char* pf = (const char*)(Wdw + o * (KTOT / 2) + (jc + 1) * (KSUB / 2));
      int h = (tid & 1) * 288;
      __builtin_prefetch(pf + h, 0, 0);
      __builtin_prefetch(pf + h + 128, 0, 0);
    }
    // ---- stage A chunk: 128 rows x 32 patch elems, expand to 9 bf16 each ----
#pragma unroll 2
    for (int it = 0; it < 8; ++it) {
      int task = it * 256 + tid;
      int jp = task & 15;        // j-pair index 0..15
      int mr = task >> 4;        // row 0..127
      int rm = rm0 + mr;         // row within batch; feature d fixed per row
      int d  = rm >> 2;
      int q  = rm & 3;
      int c  = d / KPE;
      int k9 = d - c * KPE;
      int kh = k9 / 3;
      int kw = k9 - kh * 3;
      const float* xc = xb + c * LSPAT;
      float f0[KPE], f1[KPE];
#pragma unroll
      for (int e = 0; e < 2; ++e) {
        int j  = jc * KC + jp * 2 + e;
        int l  = q * IN_DIM + j;           // spatial position index
        int oh = l / HW;
        int ow = l - oh * HW;
        int ih = oh + kh - 1;
        int iw = ow + kw - 1;
        float v = 0.0f;
        if ((unsigned)ih < (unsigned)HW && (unsigned)iw < (unsigned)HW)
          v = xc[ih * HW + iw];
        kan_feats(v, e ? f1 : f0);
      }
      unsigned int* arow = Adw + mr * LDW;
#pragma unroll
      for (int s = 0; s < KPE; ++s)
        arow[s * 16 + jp] = pack2(f0[s], f1[s]);
    }
    wait_asynccnt0();            // B chunk DMA complete (this wave's issues)
    __syncthreads();

    // ---- 9 WMMA k-steps over this chunk ----
#pragma unroll
    for (int ks = 0; ks < KPE; ++ks) {
      v16bf af[2], bfr[4];
#pragma unroll
      for (int i = 0; i < 2; ++i) {
        int mr = (wr * 2 + i) * 16 + lr;
        const unsigned int* p = Adw + mr * LDW + ks * 16 + half * 4;
        af[i] = load_frag(p, p + 8);       // K 0..7|8..15 and 16..23|24..31
      }
#pragma unroll
      for (int jn = 0; jn < 4; ++jn) {
        int n = (wc * 4 + jn) * 16 + lr;
        const unsigned int* p = Bdw + n * LDW + ks * 16 + half * 4;
        bfr[jn] = load_frag(p, p + 8);     // K 0..7|8..15 and 16..23|24..31
      }
#pragma unroll
      for (int i = 0; i < 2; ++i)
#pragma unroll
        for (int jn = 0; jn < 4; ++jn)
          acc[i][jn] = __builtin_amdgcn_wmma_f32_16x16x32_bf16(
              false, af[i], false, bfr[jn], (short)0, acc[i][jn], false, false);
    }
    __syncthreads();
  }

  // ---- scatter: out[b].flat[(m%2304)*128 + o] ----
  float* ob = out + bidx * (C_OUT * LSPAT);
#pragma unroll
  for (int i = 0; i < 2; ++i) {
#pragma unroll
    for (int jn = 0; jn < 4; ++jn) {
      int n   = (wc * 4 + jn) * 16 + lr;
      int mrb = (wr * 2 + i) * 16 + half * 8;
#pragma unroll
      for (int r = 0; r < 8; ++r) {
        int rm = rm0 + mrb + r;
        ob[rm * C_OUT + n] = acc[i][jn][r];
      }
    }
  }
}

extern "C" void kernel_launch(void* const* d_in, const int* in_sizes, int n_in,
                              void* d_out, int out_size, void* d_ws, size_t ws_size,
                              hipStream_t stream) {
  const float* x  = (const float*)d_in[0];   // (16,64,48,48)
  const float* bw = (const float*)d_in[1];   // (128,576)
  const float* sw = (const float*)d_in[2];   // (128,576,8)
  const float* sc = (const float*)d_in[3];   // (128,576)

  unsigned short* W = (unsigned short*)d_ws; // 128*5184 bf16 = 1.33 MB
  int nW = C_OUT * KTOT;
  kan_prep_w<<<(nW + 255) / 256, 256, 0, stream>>>(bw, sw, sc, W);

  size_t shmem = (size_t)(2 * MT * LDW) * sizeof(unsigned int);  // 151552 B
  int nblocks = (16 * LSPAT) / MT;  // 288
  kan_conv_wmma<<<nblocks, 256, shmem, stream>>>(x, (const unsigned int*)d_ws,
                                                 (float*)d_out);
}
